// EventDetection_34909494182492
// MI455X (gfx1250) — matrix operand order repaired and checked by reading, not compiled, not run on hardware
//
#include <hip/hip_runtime.h>
#include <math.h>

typedef unsigned short u16;
typedef unsigned int   u32;
typedef unsigned long long u64;
typedef __bf16 bf16_t;
typedef bf16_t v16bf __attribute__((ext_vector_type(16)));
typedef float  v8f   __attribute__((ext_vector_type(8)));

// ---------------- helpers ----------------
static __device__ __forceinline__ u16 f2bf(float f) {
  union { float f; u32 u; } c; c.f = f;
  u32 u = c.u;
  u += 0x7FFFu + ((u >> 16) & 1u);           // round-to-nearest-even
  return (u16)(u >> 16);
}

static __device__ __forceinline__ u32 f2bf2(float a, float b) {
  return (u32)f2bf(a) | ((u32)f2bf(b) << 16);
}

static __device__ __forceinline__ v16bf make_frag(const u16* p0, const u16* p1) {
  union { uint4 q[2]; v16bf v; } t;
  t.q[0] = *(const uint4*)p0;                // 16B LDS chunk -> ds_load_b128
  t.q[1] = *(const uint4*)p1;
  return t.v;
}

static __device__ __forceinline__ float wave_red_add(float x) {
  #pragma unroll
  for (int o = 16; o > 0; o >>= 1) x += __shfl_xor(x, o, 32);
  return x;
}

// async copy of 16B global -> LDS (CDNA5, tracked by ASYNCcnt)
static __device__ __forceinline__ void async_b128(u32 lds_off, const void* gaddr) {
  asm volatile("global_load_async_to_lds_b128 %0, %1, off"
               :: "v"(lds_off), "v"((u64)(uintptr_t)gaddr) : "memory");
}
static __device__ __forceinline__ void wait_async0() {
  asm volatile("s_wait_asynccnt 0" ::: "memory");
}

// ---------------- fp32 -> bf16 weight conversion ----------------
__global__ void cvt_bf16_k(const float* __restrict__ s, u16* __restrict__ d, int n) {
  int i = blockIdx.x * 256 + threadIdx.x;
  if (i < n) d[i] = f2bf(s[i]);
}

// ---------------- WMMA GEMM: out = A(MxK,f32) * W(NxK,bf16)^T + bias ----------------
#define BM 128
#define BN 64
#define BK 64

template <int RESID, int RELU>
__global__ __launch_bounds__(256)
void gemm_bf16(const float* __restrict__ A, const u16* __restrict__ W,
               const float* __restrict__ bias, const float* __restrict__ resid,
               float* __restrict__ out, int M, int N, int K)
{
  __shared__ __align__(16) u16 As[BM][BK];   // activations as bf16 (16KB)
  __shared__ __align__(16) u16 Bs[BN][BK];   // weights [n][k]      (8KB)

  const int tid  = threadIdx.x;
  const int wave = tid >> 5;
  const int lane = tid & 31;
  const int half = lane >> 4;
  const int l16  = lane & 15;
  const int waveM = (wave & 3) * 32;
  const int waveN = (wave >> 2) * 32;
  const int blockM = blockIdx.y * BM;
  const int blockN = blockIdx.x * BN;

  v8f acc[2][2];
  #pragma unroll
  for (int a = 0; a < 2; ++a)
    #pragma unroll
    for (int b = 0; b < 2; ++b)
      #pragma unroll
      for (int i = 0; i < 8; ++i) acc[a][b][i] = 0.0f;

  const int arow = tid >> 1;            // 2 threads per A row, 32 floats each
  const int acol = (tid & 1) * 32;
  const int brow = tid >> 2;            // 4 threads per W row, 16 bf16 each
  const int bcol = (tid & 3) * 16;

  const u32 ldsB = (u32)(uintptr_t)&Bs[brow][bcol];
  const u16* wrow = W + (size_t)(blockN + brow) * K + bcol;

  for (int k0 = 0; k0 < K; k0 += BK) {
    // B tile: async global -> LDS copy (2 x 16B), overlaps A conversion below
    async_b128(ldsB,      wrow + k0);
    async_b128(ldsB + 16, wrow + k0 + 8);

    // A tile: load 8x float4, convert f32 -> bf16, 4x ds_store_b128
    const float4* ap4 = (const float4*)(A + (size_t)(blockM + arow) * K + k0 + acol);
    float4 f[8];
    #pragma unroll
    for (int j = 0; j < 8; ++j) f[j] = ap4[j];
    union { u16 s[32]; uint4 q[4]; } t;
    const float* ff = (const float*)f;
    #pragma unroll
    for (int j = 0; j < 32; ++j) t.s[j] = f2bf(ff[j]);
    #pragma unroll
    for (int j = 0; j < 4; ++j)
      *(uint4*)&As[arow][acol + 8 * j] = t.q[j];

    wait_async0();
    __syncthreads();

    #pragma unroll
    for (int kk = 0; kk < 2; ++kk) {
      #pragma unroll
      for (int ms = 0; ms < 2; ++ms) {
        const u16* ar = &As[waveM + ms * 16 + l16][kk * 32 + half * 8];
        v16bf afrag = make_frag(ar, ar + 16);
        #pragma unroll
        for (int ns = 0; ns < 2; ++ns) {
          const u16* br = &Bs[waveN + ns * 16 + l16][kk * 32 + half * 8];
          v16bf bfrag = make_frag(br, br + 16);
          acc[ms][ns] = __builtin_amdgcn_wmma_f32_16x16x32_bf16(
              false, afrag, false, bfrag, (short)0, acc[ms][ns], false, false);
        }
      }
    }
    __syncthreads();
  }

  // straight-line epilogue (templated: no branches)
  #pragma unroll
  for (int ns = 0; ns < 2; ++ns) {
    const int col = blockN + waveN + ns * 16 + l16;
    const float bv = bias[col];
    #pragma unroll
    for (int ms = 0; ms < 2; ++ms) {
      const int row0 = blockM + waveM + ms * 16 + half * 8;
      #pragma unroll
      for (int i = 0; i < 8; ++i) {
        float v = acc[ms][ns][i] + bv;
        if (RESID) v += resid[(size_t)(row0 + i) * N + col];
        if (RELU)  v = fmaxf(v, 0.0f);
        out[(size_t)(row0 + i) * N + col] = v;
      }
    }
  }
}

// ---------------- WMMA attention, seq=16 boxes, Dh=64 ----------------
// qkv: chunk rows (512 s * 16 box) x 1536 (q|k|v); out: rows x 512
__global__ __launch_bounds__(128)
void attn16_k(const float* __restrict__ qkv, const float* __restrict__ safe,
              float* __restrict__ out, int sbase)
{
  __shared__ __align__(16) u16  qs[4][16][64];
  __shared__ __align__(16) u16  ks[4][16][64];
  __shared__ __align__(16) u16  vt[4][64][32];   // v transposed, K padded to 32
  __shared__ __align__(16) u16  at[4][16][32];   // softmax(attn), padded
  __shared__ float sc[4][16][16];

  const int wave = threadIdx.x >> 5;
  const int lane = threadIdx.x & 31;
  const int half = lane >> 4;
  const int l16  = lane & 15;
  const int item = blockIdx.x * 4 + wave;        // over 512*8 items per chunk
  const int s_local = item >> 3;
  const int h       = item & 7;

  // q/k: packed 2-wide stores (ds_store_b32)
  #pragma unroll 4
  for (int idx = lane; idx < 512; idx += 32) {
    int box = idx >> 5;
    int d2  = (idx & 31) * 2;
    const float* base = qkv + (size_t)(s_local * 16 + box) * 1536 + h * 64 + d2;
    float2 qv = *(const float2*)(base);
    float2 kv = *(const float2*)(base + 512);
    *(u32*)&qs[wave][box][d2] = f2bf2(qv.x, qv.y);
    *(u32*)&ks[wave][box][d2] = f2bf2(kv.x, kv.y);
  }
  // v transposed: pack along box axis; zero K-pad
  #pragma unroll 4
  for (int idx = lane; idx < 512; idx += 32) {
    int d  = idx >> 3;
    int b2 = (idx & 7) * 2;
    const float* vb = qkv + (size_t)(s_local * 16 + b2) * 1536 + h * 64 + 1024 + d;
    *(u32*)&vt[wave][d][b2]      = f2bf2(vb[0], vb[1536]);
    *(u32*)&vt[wave][d][16 + b2] = 0;
  }
  __syncthreads();

  // scores = q @ k^T  (K = 64 -> two 32-steps)
  v8f s0;
  #pragma unroll
  for (int i = 0; i < 8; ++i) s0[i] = 0.0f;
  #pragma unroll
  for (int kk = 0; kk < 2; ++kk) {
    const u16* ar = &qs[wave][l16][kk * 32 + half * 8];
    const u16* br = &ks[wave][l16][kk * 32 + half * 8];
    v16bf a = make_frag(ar, ar + 16);
    v16bf b = make_frag(br, br + 16);
    s0 = __builtin_amdgcn_wmma_f32_16x16x32_bf16(false, a, false, b, (short)0,
                                                 s0, false, false);
  }
  #pragma unroll
  for (int i = 0; i < 8; ++i) sc[wave][half * 8 + i][l16] = s0[i] * 0.125f;
  __syncthreads();

  if (lane < 16) {                                  // masked softmax, row = query box
    int m = lane;
    float vals[16], mx = -1e30f;
    #pragma unroll
    for (int n = 0; n < 16; ++n) {
      float v = sc[wave][m][n];
      if (safe[(size_t)(sbase + s_local) * 16 + n] == 0.0f) v = -1e9f;
      vals[n] = v; mx = fmaxf(mx, v);
    }
    float sum = 0.0f;
    #pragma unroll
    for (int n = 0; n < 16; ++n) { vals[n] = __expf(vals[n] - mx); sum += vals[n]; }
    float inv = 1.0f / sum;
    #pragma unroll
    for (int n = 0; n < 16; n += 2)
      *(u32*)&at[wave][m][n] = f2bf2(vals[n] * inv, vals[n + 1] * inv);
    #pragma unroll
    for (int n = 16; n < 32; n += 2) *(u32*)&at[wave][m][n] = 0;
  }
  __syncthreads();

  // o = attn(16x16, K padded to 32) @ v(16x64)
  {
    const u16* ar = &at[wave][l16][half * 8];
    v16bf a = make_frag(ar, ar + 16);
    #pragma unroll
    for (int ns = 0; ns < 4; ++ns) {
      const u16* br = &vt[wave][ns * 16 + l16][half * 8];
      v16bf b = make_frag(br, br + 16);
      v8f o;
      #pragma unroll
      for (int i = 0; i < 8; ++i) o[i] = 0.0f;
      o = __builtin_amdgcn_wmma_f32_16x16x32_bf16(false, a, false, b, (short)0,
                                                  o, false, false);
      #pragma unroll
      for (int i = 0; i < 8; ++i) {
        int m = half * 8 + i;
        out[(size_t)(s_local * 16 + m) * 512 + h * 64 + ns * 16 + l16] = o[i];
      }
    }
  }
}

// ---------------- seq=2 attention for fuser_ae (VALU) ----------------
__global__ __launch_bounds__(256)
void attn2_k(const float* __restrict__ qkv, float* __restrict__ out)
{
  int wave = threadIdx.x >> 5, lane = threadIdx.x & 31;
  int item = blockIdx.x * 8 + wave;      // 4096*8 items
  int s = item >> 3, h = item & 7;
  const float* b0 = qkv + (size_t)(s * 2) * 1536 + h * 64;
  const float* b1 = b0 + 1536;
  float q0a = b0[lane],        q0b = b0[lane + 32];
  float q1a = b1[lane],        q1b = b1[lane + 32];
  float k0a = b0[512 + lane],  k0b = b0[512 + lane + 32];
  float k1a = b1[512 + lane],  k1b = b1[512 + lane + 32];
  float v0a = b0[1024 + lane], v0b = b0[1024 + lane + 32];
  float v1a = b1[1024 + lane], v1b = b1[1024 + lane + 32];
  float s00 = wave_red_add(q0a * k0a + q0b * k0b) * 0.125f;
  float s01 = wave_red_add(q0a * k1a + q0b * k1b) * 0.125f;
  float s10 = wave_red_add(q1a * k0a + q1b * k0b) * 0.125f;
  float s11 = wave_red_add(q1a * k1a + q1b * k1b) * 0.125f;
  float m0 = fmaxf(s00, s01), e00 = __expf(s00 - m0), e01 = __expf(s01 - m0);
  float i0 = 1.0f / (e00 + e01), a00 = e00 * i0, a01 = e01 * i0;
  float m1 = fmaxf(s10, s11), e10 = __expf(s10 - m1), e11 = __expf(s11 - m1);
  float i1 = 1.0f / (e10 + e11), a10 = e10 * i1, a11 = e11 * i1;
  float* o0 = out + (size_t)(s * 2) * 512 + h * 64;
  float* o1 = o0 + 512;
  o0[lane]      = a00 * v0a + a01 * v1a;
  o0[lane + 32] = a00 * v0b + a01 * v1b;
  o1[lane]      = a10 * v0a + a11 * v1a;
  o1[lane + 32] = a10 * v0b + a11 * v1b;
}

// ---------------- LayerNorm (E = 512), wave per row ----------------
__global__ __launch_bounds__(256)
void layernorm_k(const float* __restrict__ x, const float* __restrict__ g,
                 const float* __restrict__ b, float* __restrict__ y, int rows)
{
  int wave = threadIdx.x >> 5, lane = threadIdx.x & 31;
  int row = blockIdx.x * 8 + wave;
  if (row >= rows) return;
  const float* xr = x + (size_t)row * 512;
  float v[16], sum = 0.0f, sq = 0.0f;
  #pragma unroll
  for (int j = 0; j < 16; ++j) {
    v[j] = xr[lane + 32 * j];
    sum += v[j]; sq += v[j] * v[j];
  }
  sum = wave_red_add(sum);
  sq  = wave_red_add(sq);
  float mu = sum * (1.0f / 512.0f);
  float var = sq * (1.0f / 512.0f) - mu * mu;
  float rstd = rsqrtf(var + 1e-5f);
  float* yr = y + (size_t)row * 512;
  #pragma unroll
  for (int j = 0; j < 16; ++j) {
    int e = lane + 32 * j;
    yr[e] = (v[j] - mu) * rstd * g[e] + b[e];
  }
}

// ---------------- fuse-prep: l2 norms, masked softmax, hard/keep/safe ----------------
__global__ __launch_bounds__(256)
void fuse_prep_k(const float* __restrict__ env, const float* __restrict__ ag,
                 const unsigned char* __restrict__ mask,
                 float* __restrict__ safe, float* __restrict__ hard,
                 float* __restrict__ keep, float* __restrict__ denom)
{
  int s = blockIdx.x, t = threadIdx.x;
  __shared__ float part[256];
  __shared__ float l2[16];
  int box = t >> 4, ln = t & 15;
  const float* e = env + (size_t)s * 512;
  const float* a = ag + ((size_t)s * 16 + box) * 512;
  float acc = 0.0f;
  #pragma unroll 4
  for (int j = 0; j < 32; ++j) {
    float x = e[ln + 16 * j] + a[ln + 16 * j];
    acc += x * x;
  }
  part[t] = acc;
  __syncthreads();
  if (ln == 0) {
    float sm = 0.0f;
    #pragma unroll
    for (int j = 0; j < 16; ++j) sm += part[box * 16 + j];
    l2[box] = sqrtf(sm);
  }
  __syncthreads();
  if (t == 0) {
    float sc[16]; int vld[16]; int nv = 0; float mx = -1e30f;
    for (int i = 0; i < 16; ++i) {
      vld[i] = (mask[(size_t)s * 16 + i] == 0);
      if (vld[i]) { nv++; sc[i] = l2[i]; } else sc[i] = -1e9f;
      mx = fmaxf(mx, sc[i]);
    }
    float sum = 0.0f;
    for (int i = 0; i < 16; ++i) { sc[i] = __expf(sc[i] - mx); sum += sc[i]; }
    float inv = 1.0f / sum;
    float th = 1.0f / fmaxf((float)nv, 1.0f);   // in (0,1]
    int cnt = 0, anyk = 0; float h[16];
    for (int i = 0; i < 16; ++i) {
      float smv = vld[i] ? sc[i] * inv : 0.0f;
      int hd = (smv >= th);
      h[i] = (float)hd; cnt += hd; anyk |= hd;
    }
    keep[s]  = anyk ? 1.0f : 0.0f;
    denom[s] = (float)((cnt > 1) ? cnt : 1);
    for (int i = 0; i < 16; ++i) {
      hard[(size_t)s * 16 + i] = h[i];
      safe[(size_t)s * 16 + i] = anyk ? h[i] : 1.0f;
    }
  }
}

// ---------------- pool over hard boxes + build pair buffer ----------------
__global__ __launch_bounds__(256)
void pool_pair_k(const float* __restrict__ fused, const float* __restrict__ env,
                 const float* __restrict__ hard, const float* __restrict__ keep,
                 const float* __restrict__ denom, float* __restrict__ pair, int sbase)
{
  int sl = blockIdx.x, s = sbase + sl, t = threadIdx.x;
  for (int e = t; e < 512; e += 256) {
    float acc = 0.0f;
    #pragma unroll
    for (int b = 0; b < 16; ++b)
      acc += hard[(size_t)s * 16 + b] * fused[((size_t)sl * 16 + b) * 512 + e];
    float pooled = (keep[s] != 0.0f) ? acc / denom[s] : 0.0f;
    pair[((size_t)s * 2 + 0) * 512 + e] = env[(size_t)s * 512 + e];
    pair[((size_t)s * 2 + 1) * 512 + e] = pooled;
  }
}

// ---------------- mean of the 2 pair tokens ----------------
__global__ void mean2_k(const float* __restrict__ src, float* __restrict__ dst, int n)
{
  int i = blockIdx.x * 256 + threadIdx.x;
  if (i >= n) return;
  int s = i >> 9, e = i & 511;
  dst[i] = 0.5f * (src[(size_t)(2 * s) * 512 + e] + src[(size_t)(2 * s + 1) * 512 + e]);
}

// ---------------- im2col for K=3 SAME conv over T=256 ----------------
__global__ void im2col3_k(const float* __restrict__ src, float* __restrict__ dst,
                          int C, int total)
{
  int idx = blockIdx.x * 256 + threadIdx.x;
  if (idx >= total) return;
  int k = idx % (C * 3);
  int r = idx / (C * 3);
  int i = k / 3, dk = k % 3;
  int t = r & 255, tt = t + dk - 1;
  dst[idx] = (tt >= 0 && tt < 256) ? src[(size_t)(r + dk - 1) * C + i] : 0.0f;
}

// ---------------- sigmoid head (N=2) ----------------
__global__ __launch_bounds__(256)
void head_k(const float* __restrict__ h2, const float* __restrict__ w,
            const float* __restrict__ b, float* __restrict__ out)
{
  int idx = blockIdx.x * 256 + threadIdx.x;          // 16*2*256
  int t = idx & 255, c = (idx >> 8) & 1, bb = idx >> 9;
  const float* x  = h2 + (size_t)(bb * 256 + t) * 256;
  const float* wc = w + c * 256;
  float acc = 0.0f;
  #pragma unroll 4
  for (int i = 0; i < 256; ++i) acc += x[i] * wc[i];
  out[(size_t)bb * 512 + c * 256 + t] = 1.0f / (1.0f + __expf(-(acc + b[c])));
}

// =============================== host ===============================
extern "C" void kernel_launch(void* const* d_in, const int* in_sizes, int n_in,
                              void* d_out, int out_size, void* d_ws, size_t ws_size,
                              hipStream_t stream)
{
  (void)in_sizes; (void)n_in; (void)out_size; (void)ws_size;
  const float* env_feat  = (const float*)d_in[0];
  const float* ag_feat   = (const float*)d_in[1];
  const unsigned char* masks = (const unsigned char*)d_in[2];
  const float* env_w = (const float*)d_in[3];
  const float* env_b = (const float*)d_in[4];
  const float* ag_w  = (const float*)d_in[5];
  const float* ag_b  = (const float*)d_in[6];
  const float* a_qkv_w = (const float*)d_in[7];   const float* a_qkv_b = (const float*)d_in[8];
  const float* a_out_w = (const float*)d_in[9];   const float* a_out_b = (const float*)d_in[10];
  const float* a_ln1g  = (const float*)d_in[11];  const float* a_ln1b  = (const float*)d_in[12];
  const float* a_f1w   = (const float*)d_in[13];  const float* a_f1b   = (const float*)d_in[14];
  const float* a_f2w   = (const float*)d_in[15];  const float* a_f2b   = (const float*)d_in[16];
  const float* a_ln2g  = (const float*)d_in[17];  const float* a_ln2b  = (const float*)d_in[18];
  const float* e_qkv_w = (const float*)d_in[19];  const float* e_qkv_b = (const float*)d_in[20];
  const float* e_out_w = (const float*)d_in[21];  const float* e_out_b = (const float*)d_in[22];
  const float* e_ln1g  = (const float*)d_in[23];  const float* e_ln1b  = (const float*)d_in[24];
  const float* e_f1w   = (const float*)d_in[25];  const float* e_f1b   = (const float*)d_in[26];
  const float* e_f2w   = (const float*)d_in[27];  const float* e_f2b   = (const float*)d_in[28];
  const float* e_ln2g  = (const float*)d_in[29];  const float* e_ln2b  = (const float*)d_in[30];
  const float* c1_w = (const float*)d_in[31];     const float* c1_b = (const float*)d_in[32];
  const float* c2_w = (const float*)d_in[33];     const float* c2_b = (const float*)d_in[34];
  const float* hd_w = (const float*)d_in[35];     const float* hd_b = (const float*)d_in[36];
  float* out = (float*)d_out;

  // ---- workspace carve ----
  char* ws = (char*)d_ws;
  size_t off = 0;
  auto take = [&](size_t bytes) -> char* {
    char* p = ws + off;
    off = (off + bytes + 255) & ~(size_t)255;
    return p;
  };
  u16* w_env  = (u16*)take((size_t)512 * 1024 * 2);
  u16* w_ag   = (u16*)take((size_t)512 * 1024 * 2);
  u16* wa_qkv = (u16*)take((size_t)1536 * 512 * 2);
  u16* wa_out = (u16*)take((size_t)512 * 512 * 2);
  u16* wa_f1  = (u16*)take((size_t)2048 * 512 * 2);
  u16* wa_f2  = (u16*)take((size_t)512 * 2048 * 2);
  u16* we_qkv = (u16*)take((size_t)1536 * 512 * 2);
  u16* we_out = (u16*)take((size_t)512 * 512 * 2);
  u16* we_f1  = (u16*)take((size_t)2048 * 512 * 2);
  u16* we_f2  = (u16*)take((size_t)512 * 2048 * 2);
  u16* w_c1   = (u16*)take((size_t)256 * 1536 * 2);
  u16* w_c2   = (u16*)take((size_t)256 * 768 * 2);

  float* env   = (float*)take((size_t)4096 * 512 * 4);
  float* ag    = (float*)take((size_t)65536 * 512 * 4);
  float* safe  = (float*)take((size_t)65536 * 4);
  float* hard  = (float*)take((size_t)65536 * 4);
  float* keep  = (float*)take((size_t)4096 * 4);
  float* denom = (float*)take((size_t)4096 * 4);
  float* Rqkv  = (float*)take((size_t)8192 * 1536 * 4);
  float* Rattn = (float*)take((size_t)8192 * 512 * 4);
  float* Rt    = (float*)take((size_t)8192 * 512 * 4);
  float* Rx1   = (float*)take((size_t)8192 * 512 * 4);
  float* Rf1   = (float*)take((size_t)8192 * 2048 * 4);
  float* Rfus  = (float*)take((size_t)8192 * 512 * 4);
  float* pair  = (float*)take((size_t)8192 * 512 * 4);

  auto cvt = [&](const float* s, u16* d, int n) {
    cvt_bf16_k<<<(n + 255) / 256, 256, 0, stream>>>(s, d, n);
  };
  auto gemm = [&](const float* A, const u16* W, const float* bias,
                  const float* resid, float* o, int M, int N, int K, int relu) {
    dim3 g(N / BN, M / BM);
    if (resid) {
      if (relu) gemm_bf16<1, 1><<<g, 256, 0, stream>>>(A, W, bias, resid, o, M, N, K);
      else      gemm_bf16<1, 0><<<g, 256, 0, stream>>>(A, W, bias, resid, o, M, N, K);
    } else {
      if (relu) gemm_bf16<0, 1><<<g, 256, 0, stream>>>(A, W, bias, resid, o, M, N, K);
      else      gemm_bf16<0, 0><<<g, 256, 0, stream>>>(A, W, bias, resid, o, M, N, K);
    }
  };

  // ---- convert all matmul weights to bf16 once ----
  cvt(env_w, w_env, 512 * 1024);   cvt(ag_w, w_ag, 512 * 1024);
  cvt(a_qkv_w, wa_qkv, 1536 * 512); cvt(a_out_w, wa_out, 512 * 512);
  cvt(a_f1w, wa_f1, 2048 * 512);    cvt(a_f2w, wa_f2, 512 * 2048);
  cvt(e_qkv_w, we_qkv, 1536 * 512); cvt(e_out_w, we_out, 512 * 512);
  cvt(e_f1w, we_f1, 2048 * 512);    cvt(e_f2w, we_f2, 512 * 2048);
  cvt(c1_w, w_c1, 256 * 1536);      cvt(c2_w, w_c2, 256 * 768);

  // ---- input linears ----
  gemm(env_feat, w_env, env_b, nullptr, env, 4096, 512, 1024, 0);
  gemm(ag_feat,  w_ag,  ag_b,  nullptr, ag, 65536, 512, 1024, 0);

  // ---- hard-attention gating masks ----
  fuse_prep_k<<<4096, 256, 0, stream>>>(env, ag, masks, safe, hard, keep, denom);

  // ---- fuser_a encoder, chunked over 8192-row (512-sequence) slices ----
  for (int c = 0; c < 8; ++c) {
    const float* agc = ag + (size_t)c * 8192 * 512;
    gemm(agc, wa_qkv, a_qkv_b, nullptr, Rqkv, 8192, 1536, 512, 0);
    attn16_k<<<1024, 128, 0, stream>>>(Rqkv, safe, Rattn, c * 512);
    gemm(Rattn, wa_out, a_out_b, agc, Rt, 8192, 512, 512, 0);
    layernorm_k<<<1024, 256, 0, stream>>>(Rt, a_ln1g, a_ln1b, Rx1, 8192);
    gemm(Rx1, wa_f1, a_f1b, nullptr, Rf1, 8192, 2048, 512, 1);
    gemm(Rf1, wa_f2, a_f2b, Rx1, Rt, 8192, 512, 2048, 0);
    layernorm_k<<<1024, 256, 0, stream>>>(Rt, a_ln2g, a_ln2b, Rfus, 8192);
    pool_pair_k<<<512, 256, 0, stream>>>(Rfus, env, hard, keep, denom, pair, c * 512);
  }

  // ---- fuser_ae encoder on (4096 x 2 x 512) pair sequences ----
  gemm(pair, we_qkv, e_qkv_b, nullptr, Rf1, 8192, 1536, 512, 0);  // qkv in Rf1
  attn2_k<<<4096, 256, 0, stream>>>(Rf1, Rattn);
  gemm(Rattn, we_out, e_out_b, pair, Rt, 8192, 512, 512, 0);
  layernorm_k<<<1024, 256, 0, stream>>>(Rt, e_ln1g, e_ln1b, Rx1, 8192);
  gemm(Rx1, we_f1, e_f1b, nullptr, Rf1, 8192, 2048, 512, 1);
  gemm(Rf1, we_f2, e_f2b, Rx1, Rt, 8192, 512, 2048, 0);
  layernorm_k<<<1024, 256, 0, stream>>>(Rt, e_ln2g, e_ln2b, Rfus, 8192);
  mean2_k<<<(4096 * 512 + 255) / 256, 256, 0, stream>>>(Rfus, env, 4096 * 512); // ctx in env

  // ---- detector: convs as im2col GEMMs ----
  im2col3_k<<<(4096 * 1536 + 255) / 256, 256, 0, stream>>>(env, Rf1, 512, 4096 * 1536);
  gemm(Rf1, w_c1, c1_b, nullptr, Rx1, 4096, 256, 1536, 1);        // h1 in Rx1
  im2col3_k<<<(4096 * 768 + 255) / 256, 256, 0, stream>>>(Rx1, Rf1, 256, 4096 * 768);
  gemm(Rf1, w_c2, c2_b, nullptr, Rattn, 4096, 256, 768, 1);       // h2 in Rattn
  head_k<<<(8192 + 255) / 256, 256, 0, stream>>>(Rattn, hd_w, hd_b, out);
}
